// Combiner_40681930227951
// MI455X (gfx1250) — matrix operand order repaired and checked by reading, not compiled, not run on hardware
//
#include <hip/hip_runtime.h>
#include <hip/hip_bf16.h>

// Problem constants (from reference): B=64, S=512, H=512, L=256
#define B_ 64
#define S_ 512
#define H_ 512
#define L_ 256
#define NWG 8            // persistent workgroups; each owns H_/NWG hc-cols, L_/NWG latent-cols
#define NTHREADS 512     // 16 waves of 32
#define HC_COLS (H_/NWG) // 64
#define L_COLS  (L_/NWG) // 32

// LDS layout (byte offsets; dynamic LDS starts at 0 — no static LDS in TU)
#define ZS_OFF    0                              // z  bf16 (B,L)    32 KB
#define HCS_OFF   (ZS_OFF   + B_*L_*2)           // hc bf16 (B,H)    64 KB
#define WC_OFF    (HCS_OFF  + B_*H_*2)           // Wc slice         32 KB
#define WMU_OFF   (WC_OFF   + HC_COLS*L_*2)      // Wmu slice        32 KB
#define WSG_OFF   (WMU_OFF  + L_COLS*H_*2)       // Wsig slice       32 KB
#define HCSL_OFF  (WSG_OFF  + L_COLS*H_*2)       // hc slice stage    8 KB
#define ZSL_OFF   (HCSL_OFF + B_*HC_COLS*2)      // z slice stage     4 KB
#define MUV_OFF   (ZSL_OFF  + B_*L_COLS*2)       // mu f32            8 KB
#define SGV_OFF   (MUV_OFF  + B_*L_COLS*4)       // sig f32           8 KB
#define SMEM_BYTES (SGV_OFF + B_*L_COLS*4)       // 225280 B ~ 220 KB

typedef __bf16 v16bf __attribute__((ext_vector_type(16)));
typedef float  v8f   __attribute__((ext_vector_type(8)));
typedef unsigned int u32x4 __attribute__((ext_vector_type(4)));
typedef int          i32x4 __attribute__((ext_vector_type(4)));
typedef int          i32x8 __attribute__((ext_vector_type(8)));

union Frag16 {
  unsigned short u16[16];
  uint4 q[2];
  v16bf bf;
};

__device__ __forceinline__ unsigned short f2bf(float f) {
  unsigned u = __float_as_uint(f);
  u += 0x7FFFu + ((u >> 16) & 1u);   // round-to-nearest-even
  return (unsigned short)(u >> 16);
}

// Branch-free tanh: (e^2x - 1)/(e^2x + 1), clamped so exp never overflows.
__device__ __forceinline__ float fast_tanh(float x) {
  x = fminf(15.0f, fmaxf(-15.0f, x));
  float e = __expf(2.0f * x);
  return (e - 1.0f) / (e + 1.0f);
}

// Branch-free softplus: max(x,0) + log(1 + e^-|x|); e^-|x| in (0,1], no overflow.
__device__ __forceinline__ float fast_softplus(float x) {
  float e = __expf(-fabsf(x));
  return fmaxf(x, 0.0f) + __logf(1.0f + e);
}

// ---- Tensor Data Mover: 2D tile copy between LDS (contiguous) and global
// (row stride).  Units are 8-byte elements (D#.data_size = 3).  D# layout per
// CDNA5 ISA ch.8: group0 {count, lds_addr, global_addr, type=2}, group1
// {data_size, tensor_dim0/1, tile_dim0/1, tensor_dim0_stride}; groups 2/3
// zero (<=2D tensor).  Completion tracked with TENSORcnt.
__device__ __forceinline__ void tdm_copy(unsigned lds_off, const void* gaddr,
                                         unsigned w8, unsigned h,
                                         unsigned stride8, bool store) {
  unsigned long long ga = (unsigned long long)gaddr;
  u32x4 g0;
  g0[0] = 1u;                                        // count=1, user mode
  g0[1] = lds_off;                                   // lds_addr [63:32]
  g0[2] = (unsigned)ga;                              // global_addr[31:0]
  g0[3] = (unsigned)((ga >> 32) & 0x1FFFFFFu)        // global_addr[56:32]
          | (2u << 30);                              // type = 2 ("image")
  i32x8 g1;
  g1[0] = (int)(3u << 16);                           // mask=0, data_size=8B
  g1[1] = (int)(w8 << 16);                           // tensor_dim0[15:0]
  g1[2] = (int)((w8 >> 16) & 0xFFFFu) | (int)(h << 16); // dim0 hi | tensor_dim1
  g1[3] = (int)((w8 & 0xFFFFu) << 16);               // tile_dim0 [127:112]
  g1[4] = (int)h;                                    // tile_dim1 (tile_dim2=0)
  g1[5] = (int)stride8;                              // tensor_dim0_stride lo32
  g1[6] = 0;
  g1[7] = 0;
  i32x4 z4 = {0, 0, 0, 0};
#if defined(__clang_major__) && (__clang_major__ >= 23)
  i32x8 z8 = {0, 0, 0, 0, 0, 0, 0, 0};
  if (store) __builtin_amdgcn_tensor_store_from_lds(g0, g1, z4, z4, z8, 0);
  else       __builtin_amdgcn_tensor_load_to_lds  (g0, g1, z4, z4, z8, 0);
#else
  if (store) __builtin_amdgcn_tensor_store_from_lds(g0, g1, z4, z4, 0);
  else       __builtin_amdgcn_tensor_load_to_lds  (g0, g1, z4, z4, 0);
#endif
}

// Device-scope barrier: fresh counter slot per barrier instance, slots zeroed
// by init kernel each launch (graph-replay deterministic).
__device__ __forceinline__ void grid_barrier(unsigned* slots, int slot) {
  __syncthreads();
  if (threadIdx.x == 0) {
    __threadfence();
    atomicAdd(&slots[slot], 1u);
    while (__hip_atomic_load(&slots[slot], __ATOMIC_ACQUIRE,
                             __HIP_MEMORY_SCOPE_AGENT) < NWG) {
      __builtin_amdgcn_s_sleep(2);
    }
  }
  __syncthreads();
}

__global__ void init_ws_kernel(unsigned* p, int n) {
  int i = blockIdx.x * blockDim.x + threadIdx.x;
  if (i < n) p[i] = 0u;
}

__global__ __launch_bounds__(NTHREADS, 1) void combiner_scan_kernel(
    const float* __restrict__ hr,    // (B,S,H)
    const float* __restrict__ eps,   // (B,S,L)
    const float* __restrict__ Wc,    // (H,L)
    const float* __restrict__ bc,    // (H)
    const float* __restrict__ Wmu,   // (L,H)
    const float* __restrict__ bmu,   // (L)
    const float* __restrict__ Wsig,  // (L,H)
    const float* __restrict__ bsig,  // (L)
    float* __restrict__ out,         // Z | mus | sigmas, each (B,S,L)
    unsigned* __restrict__ barriers, // 2*S_ counter slots
    unsigned short* __restrict__ z_g,   // (B,L) bf16 exchange buffer
    unsigned short* __restrict__ hc_g)  // (B,H) bf16 exchange buffer
{
  extern __shared__ char smem[];
  unsigned short* zs    = (unsigned short*)(smem + ZS_OFF);
  unsigned short* hcs   = (unsigned short*)(smem + HCS_OFF);
  unsigned short* Wc_s  = (unsigned short*)(smem + WC_OFF);
  unsigned short* Wmu_s = (unsigned short*)(smem + WMU_OFF);
  unsigned short* Wsg_s = (unsigned short*)(smem + WSG_OFF);
  unsigned short* hcsl  = (unsigned short*)(smem + HCSL_OFF); // (B, HC_COLS)
  unsigned short* zsl   = (unsigned short*)(smem + ZSL_OFF);  // (B, L_COLS)
  float* muv = (float*)(smem + MUV_OFF);
  float* sgv = (float*)(smem + SGV_OFF);

  const int tid   = threadIdx.x;
  const int wg    = blockIdx.x;
  const int lane  = tid & 31;
  const int wave  = tid >> 5;
  const int lhalf = lane >> 4;   // K-half select (ISA 16-bit A layout)
  const int l16   = lane & 15;

  // ---- stage weight slices as bf16 in LDS (once) ----
  for (int i = tid; i < HC_COLS*L_; i += NTHREADS) {
    int r = i >> 8, c = i & (L_-1);
    Wc_s[i] = f2bf(Wc[(wg*HC_COLS + r)*L_ + c]);
  }
  for (int i = tid; i < L_COLS*H_; i += NTHREADS) {
    int r = i >> 9, c = i & (H_-1);
    Wmu_s[i] = f2bf(Wmu [(wg*L_COLS + r)*H_ + c]);
    Wsg_s[i] = f2bf(Wsig[(wg*L_COLS + r)*H_ + c]);
  }

  // ---- static per-wave tile assignment ----
  // phase1: hc slice is 64 rows x 64 cols -> 4x4 tiles, one per wave
  const int mt1 = wave >> 2, nt1 = wave & 3;
  const int rowA1    = mt1*16 + l16;          // A-matrix row (batch)
  const int colH_loc = nt1*16 + l16;          // B/D column within slice
  const int colH     = wg*HC_COLS + colH_loc; // global hc column
  const float biasC  = bc[colH];
  const int Mbase1   = mt1*16 + lhalf*8;      // C/D row base (ISA layout)

  // phase2: waves 0..7 -> mu tiles, 8..15 -> sigma tiles; 4x2 tiles each
  const int sel = wave >> 3;
  const int mt2 = (wave & 7) >> 1, nt2 = wave & 1;
  const int rowA2    = mt2*16 + l16;
  const int colL_loc = nt2*16 + l16;
  const int colL     = wg*L_COLS + colL_loc;
  const float bias2  = sel ? bsig[colL] : bmu[colL];
  const int Mbase2   = mt2*16 + lhalf*8;
  const unsigned short* Wp = sel ? Wsg_s : Wmu_s;
  float* accout = sel ? sgv : muv;

  const long BSL = (long)B_ * S_ * L_;
  float* outZ  = out;
  float* outMu = out + BSL;
  float* outSg = out + 2*BSL;

  __syncthreads();

  for (int t = 0; t < S_; ++t) {
    // Prefetch next step's h_t tile (global_prefetch_b8)
    if (t + 1 < S_)
      __builtin_prefetch(&hr[(long)Mbase1*S_*H_ + (long)(t+1)*H_ + colH], 0, 1);

    // ---- TDM: pull full z (bf16) into LDS ----
    if (wave == 0) {
      tdm_copy(ZS_OFF, z_g, (B_*L_*2)/8, 1, (B_*L_*2)/8, false);
      __builtin_amdgcn_s_wait_tensorcnt(0);
    }
    __syncthreads();

    // ---- phase 1: hc tile = z @ Wc^T + bc, then 0.5*(tanh + h_t) ----
    {
      v8f acc;
      #pragma unroll
      for (int r = 0; r < 8; ++r) acc[r] = biasC;
      #pragma unroll
      for (int k0 = 0; k0 < L_; k0 += 32) {
        Frag16 a, b;
        const unsigned short* ap = &zs[rowA1*L_ + k0 + lhalf*8];
        a.q[0] = *(const uint4*)(ap);        // K = k0+kh*8 .. +7
        a.q[1] = *(const uint4*)(ap + 16);   // K = k0+kh*8+16 .. +23
        const unsigned short* bp = &Wc_s[colH_loc*L_ + k0 + lhalf*8];
        b.q[0] = *(const uint4*)(bp);        // K = k0+kh*8 .. +7
        b.q[1] = *(const uint4*)(bp + 16);   // K = k0+kh*8+16 .. +23
        acc = __builtin_amdgcn_wmma_f32_16x16x32_bf16(
            false, a.bf, false, b.bf, (short)0, acc, false, false);
      }
      #pragma unroll
      for (int r = 0; r < 8; ++r) {
        int m = Mbase1 + r;
        float h = __builtin_nontemporal_load(
            &hr[(long)m*S_*H_ + (long)t*H_ + colH]);
        float v = 0.5f * (fast_tanh(acc[r]) + h);
        hcsl[m*HC_COLS + colH_loc] = f2bf(v);   // stage slice for TDM store
      }
    }
    __syncthreads();
    // ---- TDM: scatter hc slice (strided rows) into global exchange buf ----
    if (wave == 0) {
      tdm_copy(HCSL_OFF, hc_g + wg*HC_COLS,
               (HC_COLS*2)/8, B_, (H_*2)/8, true);
      __builtin_amdgcn_s_wait_tensorcnt(0);
    }
    grid_barrier(barriers, 2*t);

    // ---- TDM: pull full hc (bf16) into LDS ----
    if (wave == 0) {
      tdm_copy(HCS_OFF, hc_g, (B_*H_*2)/8, 1, (B_*H_*2)/8, false);
      __builtin_amdgcn_s_wait_tensorcnt(0);
    }
    __syncthreads();

    // ---- phase 2: mu / sigma tiles = hc @ W^T + bias ----
    {
      v8f acc;
      #pragma unroll
      for (int r = 0; r < 8; ++r) acc[r] = bias2;
      #pragma unroll
      for (int k0 = 0; k0 < H_; k0 += 32) {
        Frag16 a, b;
        const unsigned short* ap = &hcs[rowA2*H_ + k0 + lhalf*8];
        a.q[0] = *(const uint4*)(ap);
        a.q[1] = *(const uint4*)(ap + 16);
        const unsigned short* bp = &Wp[colL_loc*H_ + k0 + lhalf*8];
        b.q[0] = *(const uint4*)(bp);
        b.q[1] = *(const uint4*)(bp + 16);
        acc = __builtin_amdgcn_wmma_f32_16x16x32_bf16(
            false, a.bf, false, b.bf, (short)0, acc, false, false);
      }
      #pragma unroll
      for (int r = 0; r < 8; ++r)
        accout[(Mbase2 + r)*L_COLS + colL_loc] = acc[r];
    }
    __syncthreads();

    // ---- combine: sigma = softplus, z = mu + sigma*eps; emit outputs ----
    for (int e = tid; e < B_*L_COLS; e += NTHREADS) {
      int b = e >> 5;                 // / L_COLS
      int c = e & (L_COLS - 1);
      int l = wg*L_COLS + c;
      float mu = muv[e];
      float sg = fast_softplus(sgv[e]);
      long off = (long)b*S_*L_ + (long)t*L_ + l;
      float ev = __builtin_nontemporal_load(&eps[off]);
      float z = mu + sg * ev;
      __builtin_nontemporal_store(z,  &outZ[off]);   // write-once streams:
      __builtin_nontemporal_store(mu, &outMu[off]);  // keep L2 for z/hc
      __builtin_nontemporal_store(sg, &outSg[off]);  // exchange traffic
      zsl[e] = f2bf(z);                              // stage slice for TDM
    }
    __syncthreads();
    // ---- TDM: scatter z slice (strided rows) into global exchange buf ----
    if (wave == 0) {
      tdm_copy(ZSL_OFF, z_g + wg*L_COLS,
               (L_COLS*2)/8, B_, (L_*2)/8, true);
      __builtin_amdgcn_s_wait_tensorcnt(0);
    }
    grid_barrier(barriers, 2*t + 1);
  }
}

extern "C" void kernel_launch(void* const* d_in, const int* in_sizes, int n_in,
                              void* d_out, int out_size, void* d_ws, size_t ws_size,
                              hipStream_t stream) {
  (void)in_sizes; (void)n_in; (void)out_size; (void)ws_size;
  const float* hr   = (const float*)d_in[0];
  const float* eps  = (const float*)d_in[1];
  const float* Wc   = (const float*)d_in[2];
  const float* bc   = (const float*)d_in[3];
  const float* Wmu  = (const float*)d_in[4];
  const float* bmu  = (const float*)d_in[5];
  const float* Wsig = (const float*)d_in[6];
  const float* bsig = (const float*)d_in[7];
  float* out = (float*)d_out;

  // Workspace layout: [2*S barrier slots][z bf16 (B,L)][hc bf16 (B,H)]
  unsigned char* ws = (unsigned char*)d_ws;
  unsigned*       barriers = (unsigned*)ws;                      // 4096 B
  unsigned short* z_g  = (unsigned short*)(ws + 4096);           // 32768 B
  unsigned short* hc_g = (unsigned short*)(ws + 4096 + B_*L_*2); // 65536 B

  // Zero barrier slots + z0 each launch (d_ws contents are otherwise garbage).
  const int init_words = (4096 + B_*L_*2) / 4;   // barriers + z_g
  init_ws_kernel<<<(init_words + 255) / 256, 256, 0, stream>>>(barriers, init_words);

  hipFuncSetAttribute((const void*)combiner_scan_kernel,
                      hipFuncAttributeMaxDynamicSharedMemorySize,
                      (int)SMEM_BYTES);

  combiner_scan_kernel<<<NWG, NTHREADS, SMEM_BYTES, stream>>>(
      hr, eps, Wc, bc, Wmu, bmu, Wsig, bsig, out, barriers, z_g, hc_g);
}
